// EllipticGNN_44916767981574
// MI455X (gfx1250) — matrix operand order repaired and checked by reading, not compiled, not run on hardware
//
#include <hip/hip_runtime.h>
#include <math.h>

// ---------------- constants (from reference) ----------------
static constexpr int kN     = 40000;
static constexpr int kE     = 640000;
static constexpr int kB     = 64;
static constexpr int kE2    = kE + kN;        // self-loops appended
static constexpr int kNodeF = 44;
static constexpr int kEdgeF = 95;
static constexpr int kHid   = 256;
static constexpr int kEProj = 32;

typedef __attribute__((ext_vector_type(16))) __bf16 v16bf;
typedef __attribute__((ext_vector_type(8)))  __bf16 v8bf;
typedef __attribute__((ext_vector_type(8)))  float  v8f;

// ---------------- helpers ----------------
__device__ inline __bf16 f2bf(float f) {
  unsigned u = __float_as_uint(f);
  unsigned r = (u + 0x7FFFu + ((u >> 16) & 1u)) >> 16;   // round-to-nearest-even
  unsigned short s = (unsigned short)r;
  return __builtin_bit_cast(__bf16, s);
}

__device__ inline void atomicMaxFloat(float* addr, float val) {
  if (val >= 0.0f) atomicMax((int*)addr, __float_as_int(val));
  else             atomicMin((unsigned int*)addr, __float_as_uint(val));
}

__device__ inline float waveReduceAdd(float s) {
  #pragma unroll
  for (int o = 16; o; o >>= 1) s += __shfl_xor(s, o);
  return s;
}

// ---------------- generic utility kernels ----------------
__global__ void fill_kernel(float* p, float v, long long n) {
  long long i = (long long)blockIdx.x * blockDim.x + threadIdx.x;
  if (i < n) p[i] = v;
}

// A fp32 [M,K] -> bf16 [M,Kp], zero-padded K tail
__global__ void pack_a_kernel(const float* __restrict__ src, __bf16* __restrict__ dst,
                              int M, int K, int Kp) {
  int i = blockIdx.x * blockDim.x + threadIdx.x;
  if (i >= M * Kp) return;
  int r = i / Kp, k = i - r * Kp;
  dst[i] = (k < K) ? f2bf(src[(size_t)r * K + k]) : f2bf(0.0f);
}

// W fp32 [K,N] -> bf16 transposed [N,Kp], zero-padded K tail
__global__ void pack_wt_kernel(const float* __restrict__ src, __bf16* __restrict__ dst,
                               int K, int N, int Kp) {
  int i = blockIdx.x * blockDim.x + threadIdx.x;
  if (i >= N * Kp) return;
  int n = i / Kp, k = i - n * Kp;
  dst[i] = (k < K) ? f2bf(src[(size_t)k * N + n]) : f2bf(0.0f);
}

// ---------------- WMMA GEMM ------------------------------------------------
// C[M,N] = act(A_bf16[M,Kp] @ Bt_bf16[N,Kp]^T + bias)
// Per wave: one 16x32 C macro-tile (2 WMMA tiles sharing the A fragment).
// Fragment layouts used here (contiguous k order, not the ISA table order):
//   A 16x32: lane m=lane&15; lane<16 -> elems 0..15=K 0..15;
//            lane>=16 -> elems 0..15=K 16..31.
//   B 32x16: lane n=lane&15; lane<16 -> elems 0..15=K 0..15; lane>=16 -> K 16..31.
// Packed operands make every fragment two contiguous 16B loads per lane.
__global__ void wmma_gemm_kernel(const __bf16* __restrict__ A, const __bf16* __restrict__ Bt,
                                 const float* __restrict__ bias, float* __restrict__ C,
                                 int M, int N, int Kp, int act) {
  int wid  = threadIdx.x >> 5;
  int lane = threadIdx.x & 31;
  int mtile = blockIdx.x * 4 + wid;            // 4 waves / block
  if (mtile * 16 >= M) return;                 // wave-uniform: EXEC stays all-ones
  int row0 = mtile * 16;
  int col0 = blockIdx.y * 32;
  int ln15 = lane & 15;
  bool hiHalf = (lane >= 16);

  const __bf16* pa  = A  + (size_t)(row0 + ln15) * Kp + (hiHalf ? 16 : 0);
  const __bf16* pb0 = Bt + (size_t)(col0 + ln15) * Kp + (hiHalf ? 16 : 0);
  const __bf16* pb1 = pb0 + (size_t)16 * Kp;

  v8f acc0 = {}, acc1 = {};
  for (int k0 = 0; k0 < Kp; k0 += 32) {
    __builtin_prefetch(pa + k0 + 128, 0, 1);     // speculative: global_prefetch_b8
    v8bf alo = *(const v8bf*)(pa  + k0);
    v8bf ahi = *(const v8bf*)(pa  + k0 + 8);
    v8bf x0  = *(const v8bf*)(pb0 + k0);
    v8bf x1  = *(const v8bf*)(pb0 + k0 + 8);
    v8bf y0  = *(const v8bf*)(pb1 + k0);
    v8bf y1  = *(const v8bf*)(pb1 + k0 + 8);
    v16bf a, b0, b1;
    #pragma unroll
    for (int i = 0; i < 8; ++i) {
      a[i] = alo[i];  a[8 + i] = ahi[i];
      b0[i] = x0[i];  b0[8 + i] = x1[i];
      b1[i] = y0[i];  b1[8 + i] = y1[i];
    }
    acc0 = __builtin_amdgcn_wmma_f32_16x16x32_bf16(false, a, false, b0, (short)0, acc0, false, false);
    acc1 = __builtin_amdgcn_wmma_f32_16x16x32_bf16(false, a, false, b1, (short)0, acc1, false, false);
  }

  // C/D layout: vgpr r -> row = r + 8*(lane>=16), col = lane&15
  int mb = hiHalf ? 8 : 0;
  #pragma unroll
  for (int t = 0; t < 2; ++t) {
    int col = col0 + t * 16 + ln15;
    float bv = bias ? bias[col] : 0.0f;
    v8f acc = t ? acc1 : acc0;
    #pragma unroll
    for (int r = 0; r < 8; ++r) {
      float v = acc[r] + bv;
      if (act == 1)      v = fmaxf(v, 0.0f);
      else if (act == 2) v = tanhf(v);
      C[(size_t)(row0 + mb + r) * N + col] = v;
    }
  }
}

// ---------------- LayerNorm (+bias) -> ReLU (+residual), one wave per row -----
__global__ void ln_relu_kernel(const float* __restrict__ in, float* __restrict__ out,
                               const float* __restrict__ g, const float* __restrict__ be,
                               const float* __restrict__ addbias,
                               const float* __restrict__ resid, int rows, int D) {
  int wid  = threadIdx.x >> 5;
  int lane = threadIdx.x & 31;
  int row  = blockIdx.x * (blockDim.x >> 5) + wid;
  if (row >= rows) return;
  int per = D >> 5;                // elements per lane (<=8)
  float v[8];
  float s = 0.0f;
  for (int c = 0; c < per; ++c) {
    int j = c * 32 + lane;
    float xv = in[(size_t)row * D + j];
    if (addbias) xv += addbias[j];
    v[c] = xv;  s += xv;
  }
  s = waveReduceAdd(s);
  float mu = s / (float)D;
  float q = 0.0f;
  for (int c = 0; c < per; ++c) { float d = v[c] - mu; q += d * d; }
  q = waveReduceAdd(q);
  float rstd = rsqrtf(q / (float)D + 1e-5f);
  for (int c = 0; c < per; ++c) {
    int j = c * 32 + lane;
    float y = (v[c] - mu) * rstd * g[j] + be[j];
    y = fmaxf(y, 0.0f);
    size_t idx = (size_t)row * D + j;
    out[idx] = resid ? (resid[idx] + y) : y;
  }
}

// ---------------- edge feature mean (for self-loop fill) ----------------------
__global__ void col_sum32_kernel(const float* __restrict__ ea, float* __restrict__ colsum, int E) {
  int t = blockIdx.x * blockDim.x + threadIdx.x;
  int col = t & 31;
  int r0  = t >> 5;
  int stride = (gridDim.x * blockDim.x) >> 5;
  float s = 0.0f;
  for (int r = r0; r < E; r += stride) s += ea[(size_t)r * 32 + col];
  atomicAdd(&colsum[col], s);
}

__global__ void fill_mean_rows_kernel(const float* __restrict__ colsum,
                                      float* __restrict__ ea2, int E, int n) {
  int t = blockIdx.x * blockDim.x + threadIdx.x;
  if (t >= n * 32) return;
  int c = t & 31;
  ea2[(size_t)(E + (t >> 5)) * 32 + c] = colsum[c] / (float)E;
}

// ---------------- GATv2 edge logits (ef recomputed on the fly) ----------------
__global__ void edge_logits_kernel(const float* __restrict__ ea2, const float* __restrict__ We,
                                   const float* __restrict__ xl, const float* __restrict__ xr,
                                   const float* __restrict__ att, const int* __restrict__ ei,
                                   float* __restrict__ logit, int E, int E2) {
  int wid  = threadIdx.x >> 5;
  int lane = threadIdx.x & 31;
  int e = blockIdx.x * 8 + wid;
  if (e >= E2) return;
  int src = (e < E) ? ei[e]     : (e - E);
  int dst = (e < E) ? ei[E + e] : (e - E);
  float ea[32];
  #pragma unroll
  for (int k = 0; k < 32; ++k) ea[k] = ea2[(size_t)e * 32 + k];   // wave-uniform loads

  #pragma unroll
  for (int h = 0; h < 4; ++h) {
    float s = 0.0f;
    #pragma unroll
    for (int half = 0; half < 2; ++half) {
      int d = half * 32 + lane;
      int j = h * 64 + d;
      float ef = 0.0f;
      #pragma unroll
      for (int k = 0; k < 32; ++k) ef = fmaf(ea[k], We[k * 256 + j], ef);
      float z = xl[(size_t)src * 256 + j] + xr[(size_t)dst * 256 + j] + ef;
      z = (z > 0.0f) ? z : 0.2f * z;                               // leaky_relu
      s = fmaf(att[h * 64 + d], z, s);
    }
    s = waveReduceAdd(s);
    if (lane == 0) logit[(size_t)e * 4 + h] = s;
  }
}

__global__ void seg_max_kernel(const float* __restrict__ logit, const int* __restrict__ ei,
                               float* __restrict__ m, int E, int E2) {
  int t = blockIdx.x * blockDim.x + threadIdx.x;
  if (t >= E2 * 4) return;
  int e = t >> 2, h = t & 3;
  int dst = (e < E) ? ei[E + e] : (e - E);
  atomicMaxFloat(&m[dst * 4 + h], logit[t]);
}

__global__ void wden_kernel(float* __restrict__ logit, const float* __restrict__ m,
                            float* __restrict__ den, const int* __restrict__ ei, int E, int E2) {
  int t = blockIdx.x * blockDim.x + threadIdx.x;
  if (t >= E2 * 4) return;
  int e = t >> 2, h = t & 3;
  int dst = (e < E) ? ei[E + e] : (e - E);
  float w = expf(logit[t] - m[dst * 4 + h]);
  logit[t] = w;                       // buffer reused to hold w
  atomicAdd(&den[dst * 4 + h], w);
}

__global__ void agg_kernel(const float* __restrict__ xl, const float* __restrict__ w,
                           const float* __restrict__ den, const int* __restrict__ ei,
                           float* __restrict__ agg, int E) {
  int e = blockIdx.x;
  int j = threadIdx.x;                // 256 threads = full feature row
  int h = j >> 6;
  int src = (e < E) ? ei[e]     : (e - E);
  int dst = (e < E) ? ei[E + e] : (e - E);
  float alpha = w[(size_t)e * 4 + h] / (den[dst * 4 + h] + 1e-16f);
  atomicAdd(&agg[(size_t)dst * 256 + j], alpha * xl[(size_t)src * 256 + j]);
}

// ---------------- attention pooling ----------------
__global__ void pool_scores_kernel(const float* __restrict__ t1, const float* __restrict__ W2,
                                   const float* __restrict__ b2, const int* __restrict__ batch,
                                   float* __restrict__ scores, float* __restrict__ ssum, int n) {
  int wid  = threadIdx.x >> 5;
  int lane = threadIdx.x & 31;
  int node = blockIdx.x * 8 + wid;
  if (node >= n) return;
  float s = 0.0f;
  #pragma unroll
  for (int c = 0; c < 4; ++c) {
    int j = c * 32 + lane;
    s = fmaf(t1[(size_t)node * 128 + j], W2[j], s);
  }
  s = waveReduceAdd(s);
  if (lane == 0) {
    s += b2[0];
    scores[node] = s;
    atomicAdd(&ssum[batch[node]], s);
  }
}

__global__ void pool_wexp_kernel(float* __restrict__ scores, const float* __restrict__ ssum,
                                 float* __restrict__ wsum, const int* __restrict__ batch, int n) {
  int i = blockIdx.x * blockDim.x + threadIdx.x;
  if (i >= n) return;
  int b = batch[i];
  float w = expf(scores[i] - ssum[b]);
  scores[i] = w;
  atomicAdd(&wsum[b], w);
}

__global__ void pool_acc_kernel(const float* __restrict__ h, const float* __restrict__ wbuf,
                                const float* __restrict__ wsum, const int* __restrict__ batch,
                                float* __restrict__ xatt, float* __restrict__ xsum,
                                float* __restrict__ xmax, float* __restrict__ cnt) {
  int node = blockIdx.x;
  int j = threadIdx.x;                // 256
  int b = batch[node];
  float wn = wbuf[node] / (wsum[b] + 1e-8f);
  float hv = h[(size_t)node * 256 + j];
  atomicAdd(&xatt[(size_t)b * 256 + j], wn * hv);
  atomicAdd(&xsum[(size_t)b * 256 + j], hv);
  atomicMaxFloat(&xmax[(size_t)b * 256 + j], hv);
  if (j == 0) atomicAdd(&cnt[b], 1.0f);
}

__global__ void pooled_build_kernel(const float* __restrict__ xatt, const float* __restrict__ xsum,
                                    const float* __restrict__ xmax, const float* __restrict__ cnt,
                                    float* __restrict__ pooled) {
  int t = blockIdx.x * blockDim.x + threadIdx.x;
  if (t >= kB * 768) return;
  int b = t / 768, j = t % 768;
  float v;
  if (j < 256)       v = xatt[b * 256 + j];
  else if (j < 512)  v = xsum[b * 256 + (j - 256)] / fmaxf(cnt[b], 1.0f);
  else               v = xmax[b * 256 + (j - 512)];
  pooled[t] = v;
}

// ---------------- classifier epilogues ----------------
__global__ void bn_relu_kernel(float* __restrict__ y, const float* __restrict__ g,
                               const float* __restrict__ be, int total, int D) {
  int i = blockIdx.x * blockDim.x + threadIdx.x;
  if (i >= total) return;
  int c = i % D;
  float rs = rsqrtf(1.0f + 1e-5f);
  y[i] = fmaxf(y[i] * rs * g[c] + be[c], 0.0f);
}

__global__ void clf_final_kernel(const float* __restrict__ y3, const float* __restrict__ W4,
                                 const float* __restrict__ b4, float* __restrict__ out) {
  int t = blockIdx.x * blockDim.x + threadIdx.x;
  if (t >= kB * 2) return;
  int b = t >> 1, c = t & 1;
  float s = b4[c];
  #pragma unroll
  for (int k = 0; k < 64; ++k) s = fmaf(y3[b * 64 + k], W4[k * 2 + c], s);
  out[t] = s;
}

// =====================================================================
extern "C" void kernel_launch(void* const* d_in, const int* in_sizes, int n_in,
                              void* d_out, int out_size, void* d_ws, size_t ws_size,
                              hipStream_t stream) {
  (void)in_sizes; (void)n_in; (void)out_size; (void)ws_size;
  auto F = [&](int i) { return (const float*)d_in[i]; };

  const float* x     = F(0);
  const int*   ei    = (const int*)d_in[1];
  const float* eattr = F(2);
  const int*   batch = (const int*)d_in[3];
  const float *epW = F(4),  *epb = F(5),  *epg = F(6),  *epbe = F(7);
  const float *ipW = F(8),  *ipb = F(9),  *ipg = F(10), *ipbe = F(11);
  // gat layers at 12 + 9*l : Wl,bl,Wr,br,We,att,bias,g,be
  const float *pW1 = F(39), *pb1 = F(40), *pW2 = F(41), *pb2 = F(42);
  const float *cW1 = F(43), *cb1 = F(44), *cg1 = F(45), *cbe1 = F(46);
  const float *cW2 = F(47), *cb2 = F(48), *cg2 = F(49), *cbe2 = F(50);
  const float *cW3 = F(51), *cb3 = F(52), *cg3 = F(53), *cbe3 = F(54);
  const float *cW4 = F(55), *cb4 = F(56);

  // ---- workspace carve-out ----
  char* wp = (char*)d_ws;
  auto carve = [&](size_t bytes) { void* p = wp; wp += (bytes + 255) & ~size_t(255); return p; };
  float* ea2    = (float*)carve((size_t)kE2 * 32 * 4);
  float* h      = (float*)carve((size_t)kN * 256 * 4);
  float* xl     = (float*)carve((size_t)kN * 256 * 4);
  float* xr     = (float*)carve((size_t)kN * 256 * 4);
  float* agg    = (float*)carve((size_t)kN * 256 * 4);
  float* t1     = (float*)carve((size_t)kN * 128 * 4);
  float* lw     = (float*)carve((size_t)kE2 * 4 * 4);
  float* mbuf   = (float*)carve((size_t)kN * 4 * 4);
  float* den    = (float*)carve((size_t)kN * 4 * 4);
  float* scores = (float*)carve((size_t)kN * 4);
  float* colsum = (float*)carve(32 * 4);
  float* ssum   = (float*)carve(kB * 4);
  float* wsum   = (float*)carve(kB * 4);
  float* cnt    = (float*)carve(kB * 4);
  float* xatt   = (float*)carve((size_t)kB * 256 * 4);
  float* xsum   = (float*)carve((size_t)kB * 256 * 4);
  float* xmax   = (float*)carve((size_t)kB * 256 * 4);
  float* pooled = (float*)carve((size_t)kB * 768 * 4);
  float* y1     = (float*)carve((size_t)kB * 256 * 4);
  float* y2     = (float*)carve((size_t)kB * 128 * 4);
  float* y3     = (float*)carve((size_t)kB * 64 * 4);
  __bf16* Abf   = (__bf16*)carve((size_t)kE * 96 * 2);     // largest packed A (edge_attr)
  __bf16* wbf   = (__bf16*)carve((size_t)256 * 768 * 2);   // largest packed W^T

  const float NEGINF = -INFINITY;
  auto fill = [&](float* p, float v, size_t n) {
    fill_kernel<<<dim3((unsigned)((n + 255) / 256)), dim3(256), 0, stream>>>(p, v, (long long)n);
  };
  auto rup32 = [](int k) { return (k + 31) & ~31; };
  auto packA = [&](const float* A, int M, int K) {
    int Kp = rup32(K);
    int n = M * Kp;
    pack_a_kernel<<<(n + 255) / 256, 256, 0, stream>>>(A, Abf, M, K, Kp);
    return Kp;
  };
  auto gemm = [&](const float* Wf, const float* bias, float* C,
                  int M, int N, int K, int Kp, int act) {
    int wn = N * Kp;
    pack_wt_kernel<<<(wn + 255) / 256, 256, 0, stream>>>(Wf, wbf, K, N, Kp);
    dim3 grid((M / 16 + 3) / 4, N / 32);
    wmma_gemm_kernel<<<grid, dim3(128), 0, stream>>>(Abf, wbf, bias, C, M, N, Kp, act);
  };
  auto ln = [&](const float* in, float* out, const float* g, const float* be,
                const float* addbias, const float* resid, int rows, int D) {
    ln_relu_kernel<<<(rows + 7) / 8, 256, 0, stream>>>(in, out, g, be, addbias, resid, rows, D);
  };

  int Kp;

  // ---- edge projection: Linear -> LN -> ReLU, into ea2[0..E) ----
  Kp = packA(eattr, kE, kEdgeF);
  gemm(epW, epb, ea2, kE, kEProj, kEdgeF, Kp, 0);
  ln(ea2, ea2, epg, epbe, nullptr, nullptr, kE, kEProj);
  fill(colsum, 0.0f, 32);
  col_sum32_kernel<<<1024, 256, 0, stream>>>(ea2, colsum, kE);
  fill_mean_rows_kernel<<<(kN * 32 + 255) / 256, 256, 0, stream>>>(colsum, ea2, kE, kN);

  // ---- input projection: Linear -> LN -> ReLU ----
  Kp = packA(x, kN, kNodeF);
  gemm(ipW, ipb, h, kN, kHid, kNodeF, Kp, 0);
  ln(h, h, ipg, ipbe, nullptr, nullptr, kN, kHid);

  // ---- 3 GATv2 layers ----
  for (int l = 0; l < 3; ++l) {
    int base = 12 + l * 9;
    const float *Wl = F(base + 0), *bl = F(base + 1);
    const float *Wr = F(base + 2), *br = F(base + 3);
    const float *We = F(base + 4), *att = F(base + 5);
    const float *gb = F(base + 6), *gg = F(base + 7), *gbe = F(base + 8);

    Kp = packA(h, kN, kHid);              // pack h once, reuse for both GEMMs
    gemm(Wl, bl, xl, kN, kHid, kHid, Kp, 0);
    gemm(Wr, br, xr, kN, kHid, kHid, Kp, 0);

    edge_logits_kernel<<<(kE2 + 7) / 8, 256, 0, stream>>>(ea2, We, xl, xr, att, ei, lw, kE, kE2);

    fill(mbuf, NEGINF, (size_t)kN * 4);
    fill(den, 0.0f, (size_t)kN * 4);
    fill(agg, 0.0f, (size_t)kN * 256);

    seg_max_kernel<<<(kE2 * 4 + 255) / 256, 256, 0, stream>>>(lw, ei, mbuf, kE, kE2);
    wden_kernel<<<(kE2 * 4 + 255) / 256, 256, 0, stream>>>(lw, mbuf, den, ei, kE, kE2);
    agg_kernel<<<kE2, 256, 0, stream>>>(xl, lw, den, ei, agg, kE);

    // out = relu(LN(agg + bias)); h = h + out
    ln(agg, h, gg, gbe, gb, h, kN, kHid);
  }

  // ---- attention pooling ----
  Kp = packA(h, kN, kHid);
  gemm(pW1, pb1, t1, kN, 128, kHid, Kp, 2);   // tanh
  fill(ssum, 0.0f, kB); fill(wsum, 0.0f, kB); fill(cnt, 0.0f, kB);
  fill(xatt, 0.0f, (size_t)kB * 256); fill(xsum, 0.0f, (size_t)kB * 256);
  fill(xmax, NEGINF, (size_t)kB * 256);
  pool_scores_kernel<<<(kN + 7) / 8, 256, 0, stream>>>(t1, pW2, pb2, batch, scores, ssum, kN);
  pool_wexp_kernel<<<(kN + 255) / 256, 256, 0, stream>>>(scores, ssum, wsum, batch, kN);
  pool_acc_kernel<<<kN, 256, 0, stream>>>(h, scores, wsum, batch, xatt, xsum, xmax, cnt);
  pooled_build_kernel<<<(kB * 768 + 255) / 256, 256, 0, stream>>>(xatt, xsum, xmax, cnt, pooled);

  // ---- classifier ----
  Kp = packA(pooled, kB, 768);
  gemm(cW1, cb1, y1, kB, 256, 768, Kp, 0);
  bn_relu_kernel<<<(kB * 256 + 255) / 256, 256, 0, stream>>>(y1, cg1, cbe1, kB * 256, 256);
  Kp = packA(y1, kB, 256);
  gemm(cW2, cb2, y2, kB, 128, 256, Kp, 0);
  bn_relu_kernel<<<(kB * 128 + 255) / 256, 256, 0, stream>>>(y2, cg2, cbe2, kB * 128, 128);
  Kp = packA(y2, kB, 128);
  gemm(cW3, cb3, y3, kB, 64, 128, Kp, 0);
  bn_relu_kernel<<<(kB * 64 + 255) / 256, 256, 0, stream>>>(y3, cg3, cbe3, kB * 64, 64);
  clf_final_kernel<<<1, 128, 0, stream>>>(y3, cW4, cb4, (float*)d_out);
}
